// Decoder_39393440039096
// MI455X (gfx1250) — matrix operand order repaired and checked by reading, not compiled, not run on hardware
//
#include <hip/hip_runtime.h>
#include <hip/hip_bf16.h>
#include <math.h>

typedef __bf16 bf16_t;
typedef __attribute__((ext_vector_type(16))) __bf16        v16bf;
typedef __attribute__((ext_vector_type(8)))  float         v8f;
typedef __attribute__((ext_vector_type(4)))  unsigned int  u32x4;
typedef __attribute__((ext_vector_type(4)))  float         f32x4;
typedef __attribute__((ext_vector_type(4)))  int           v4i;

// Address-space-qualified pointee types for the async global->LDS builtin
// (its prototype takes `int4 __device__*` / `int4 __shared__*`).
typedef __attribute__((address_space(1))) v4i as1_v4i;
typedef __attribute__((address_space(3))) v4i as3_v4i;

constexpr int B = 128, T = 48, S = 256, ENC_D = 512;
constexpr int U = 256, ATTN_U = 128, EMB_D = 256;

#if __has_builtin(__builtin_amdgcn_global_load_async_to_lds_b128)
#define HAVE_ASYNC_LDS 1
#else
#define HAVE_ASYNC_LDS 0
#endif

__device__ __forceinline__ void async_wait0() {
#if __has_builtin(__builtin_amdgcn_s_wait_asynccnt)
    __builtin_amdgcn_s_wait_asynccnt(0);
#else
    asm volatile("s_wait_asynccnt 0" ::: "memory");
#endif
}

__device__ __forceinline__ float fast_tanh(float x) {
#if __has_builtin(__builtin_amdgcn_tanhf)
    return __builtin_amdgcn_tanhf(x);
#else
    return tanhf(x);
#endif
}

__device__ __forceinline__ float sigmoidf_(float x) { return 1.0f / (1.0f + __expf(-x)); }

__device__ __forceinline__ v8f vzero8() {
    v8f z;
#pragma unroll
    for (int i = 0; i < 8; ++i) z[i] = 0.0f;
    return z;
}

union AFrag { v16bf v; u32x4 q[2]; };

// A fragment (16x32 bf16, row-major source with leading dim ld).
// ISA layout: lane m=L&15, half=L>>4; elems 0..7 -> K = half*8+e, elems 8..15 -> K = 16+half*8+(e-8).
__device__ __forceinline__ v16bf load_a_bf16(const bf16_t* __restrict__ src, int ld,
                                             int row0, int k0, int lane) {
    int m = lane & 15, half = lane >> 4;
    const bf16_t* p = src + (size_t)(row0 + m) * ld + k0 + half * 8;
    AFrag a;
    a.q[0] = *(const u32x4*)(p);
    a.q[1] = *(const u32x4*)(p + 16);
    return a.v;
}

// A fragment from fp32 row-major source, converted to bf16 in-register (v_cvt_pk_bf16_f32).
__device__ __forceinline__ v16bf load_a_f32(const float* __restrict__ src, int ld,
                                            int row0, int k0, int lane) {
    int m = lane & 15, half = lane >> 4;
    const float* p = src + (size_t)(row0 + m) * ld + k0 + half * 8;
    f32x4 f0 = *(const f32x4*)(p);
    f32x4 f1 = *(const f32x4*)(p + 4);
    f32x4 f2 = *(const f32x4*)(p + 16);
    f32x4 f3 = *(const f32x4*)(p + 20);
    v16bf a;
#pragma unroll
    for (int i = 0; i < 4; ++i) {
        a[i]      = (bf16_t)f0[i];
        a[4 + i]  = (bf16_t)f1[i];
        a[8 + i]  = (bf16_t)f2[i];
        a[12 + i] = (bf16_t)f3[i];
    }
    return a;
}

// Pre-packed B fragment: one contiguous 32B load per lane (1KB contiguous per wave-tile).
__device__ __forceinline__ v16bf load_b_packed(const bf16_t* __restrict__ packed,
                                               int NT, int kt, int nt, int lane) {
    return *(const v16bf*)(packed + ((((size_t)kt * NT + nt) * 32) + lane) * 16);
}

__device__ __forceinline__ v8f wmma_bf16(v16bf a, v16bf b, v8f c) {
    return __builtin_amdgcn_wmma_f32_16x16x32_bf16(false, a, false, b, (short)0, c, false, false);
}

// ---- Prep kernels -----------------------------------------------------------

// Pack [K,N] fp32 (optionally stacked src0[0:ksplit) / src1[ksplit:...)) into bf16 WMMA B-tile order.
// B-layout per 32x16 tile: lane L holds N = nt*16 + (L&15); element e holds K = kt*32 + (L>>4)*8 + e (e<8), kt*32 + 16 + (L>>4)*8 + (e-8) (e>=8).
__global__ void pack_b_kernel(const float* __restrict__ src0, const float* __restrict__ src1,
                              int ksplit, int ld, bf16_t* __restrict__ dst, int KT, int NT) {
    int t = blockIdx.x * blockDim.x + threadIdx.x;
    int total = KT * NT * 32;
    if (t >= total) return;
    int lane = t & 31;
    int tile = t >> 5;
    int kt = tile / NT, nt = tile % NT;
    int n = nt * 16 + (lane & 15);
    int kbase = kt * 32 + (lane >> 4) * 8;
    bf16_t* d = dst + (size_t)t * 16;
#pragma unroll
    for (int e = 0; e < 16; ++e) {
        int k = kbase + (e < 8 ? e : e + 8);
        float v = (k < ksplit) ? src0[(size_t)k * ld + n] : src1[(size_t)(k - ksplit) * ld + n];
        d[e] = (bf16_t)v;
    }
}

// emb[t,b,u] = relu(x[b,t] * embW[u] + embB[u]) as bf16 (no recurrence -> hoisted for all t)
__global__ void emb_kernel(const float* __restrict__ x, const float* __restrict__ embW,
                           const float* __restrict__ embB, bf16_t* __restrict__ out) {
    int idx = blockIdx.x * blockDim.x + threadIdx.x;
    if (idx >= T * B * EMB_D) return;
    int u = idx & (EMB_D - 1);
    int b = (idx >> 8) & (B - 1);
    int t = idx >> 15;
    float v = x[b * T + t] * embW[u] + embB[u];
    out[idx] = (bf16_t)fmaxf(v, 0.0f);
}

// Initialize recurrent state copies (inputs must not be mutated).
__global__ void init_state_kernel(const float* __restrict__ h1, const float* __restrict__ c1,
                                  const float* __restrict__ h2, const float* __restrict__ c2,
                                  bf16_t* __restrict__ cxh1_0, bf16_t* __restrict__ in2_0,
                                  float* __restrict__ c1s, float* __restrict__ c2s) {
    int idx = blockIdx.x * blockDim.x + threadIdx.x;
    if (idx >= B * U) return;
    int b = idx >> 8, u = idx & 255;
    cxh1_0[(size_t)b * 768 + 512 + u] = (bf16_t)h1[idx];   // h1 lives in cols [512,768) of [ctx|h1]
    in2_0[(size_t)b * 512 + 256 + u] = (bf16_t)h2[idx];    // h2 lives in cols [256,512) of [h1n|h2]
    c1s[idx] = c1[idx];
    c2s[idx] = c2[idx];
}

// keys = enc @ W1 + b1 : [32768,512] x [512,128], WMMA bf16. Block = 16-row M tile, wave = N tile.
__global__ void keys_kernel(const float* __restrict__ enc, const bf16_t* __restrict__ W1p,
                            const float* __restrict__ b1, float* __restrict__ keys) {
    int lane = threadIdx.x & 31;
    int wave = threadIdx.x >> 5;   // N-tile 0..7
    int row0 = blockIdx.x * 16;    // 2048 blocks
    v8f acc = vzero8();
#pragma unroll 4
    for (int kt = 0; kt < 16; ++kt) {
        v16bf a = load_a_f32(enc, ENC_D, row0, kt * 32, lane);
        v16bf b = load_b_packed(W1p, 8, kt, wave, lane);
        acc = wmma_bf16(a, b, acc);
    }
    int n = wave * 16 + (lane & 15);
    float bias = b1[n];
    int rbase = row0 + ((lane >> 4) ? 8 : 0);
#pragma unroll
    for (int r = 0; r < 8; ++r)
        keys[(size_t)(rbase + r) * ATTN_U + n] = acc[r] + bias;
}

// ---- Per-step kernels -------------------------------------------------------

// q = h2 @ W2 + b2 : [128,256] x [256,128]
__global__ void qgemm_kernel(const bf16_t* __restrict__ h2bf, const bf16_t* __restrict__ W2p,
                             const float* __restrict__ b2, float* __restrict__ q) {
    int lane = threadIdx.x & 31;
    int wave = threadIdx.x >> 5;   // N-tile 0..7
    int row0 = blockIdx.x * 16;    // 8 blocks
    v8f acc = vzero8();
#pragma unroll
    for (int kt = 0; kt < 8; ++kt) {
        v16bf a = load_a_bf16(h2bf, 512, row0, kt * 32, lane);   // h2 region of in2, ld=512
        v16bf b = load_b_packed(W2p, 8, kt, wave, lane);
        acc = wmma_bf16(a, b, acc);
    }
    int n = wave * 16 + (lane & 15);
    float bias = b2[n];
    int rbase = row0 + ((lane >> 4) ? 8 : 0);
#pragma unroll
    for (int r = 0; r < 8; ++r)
        q[(size_t)(rbase + r) * ATTN_U + n] = acc[r] + bias;
}

// Fused Bahdanau score + softmax + context. One block per batch row; thread = source position s.
__global__ void attn_kernel(const float* __restrict__ keys, const float* __restrict__ q,
                            const float* __restrict__ V, const float* __restrict__ bV,
                            const float* __restrict__ enc, bf16_t* __restrict__ cxh1_cur) {
    __shared__ float qs[ATTN_U];
    __shared__ float sc[S];
    __shared__ float red[256];
    int b = blockIdx.x, tid = threadIdx.x;
    if (tid < ATTN_U) qs[tid] = q[b * ATTN_U + tid];
    __syncthreads();

    const float* krow = keys + ((size_t)b * S + tid) * ATTN_U;
    float acc = 0.0f;
#pragma unroll 8
    for (int u = 0; u < ATTN_U; ++u) acc += fast_tanh(krow[u] + qs[u]) * V[u];
    acc += bV[0];

    red[tid] = acc;
    __syncthreads();
    for (int off = 128; off > 0; off >>= 1) {
        if (tid < off) red[tid] = fmaxf(red[tid], red[tid + off]);
        __syncthreads();
    }
    float mx = red[0];
    __syncthreads();
    float e = __expf(acc - mx);
    sc[tid] = e;
    red[tid] = e;
    __syncthreads();
    for (int off = 128; off > 0; off >>= 1) {
        if (tid < off) red[tid] += red[tid + off];
        __syncthreads();
    }
    float inv = 1.0f / red[0];
    __syncthreads();

    // ctx[b, tid] and ctx[b, tid+256]; coalesced over d, attention weights from LDS
    float a0 = 0.0f, a1 = 0.0f;
    const float* eb = enc + (size_t)b * S * ENC_D;
    for (int s = 0; s < S; ++s) {
        float w = sc[s] * inv;
        a0 += w * eb[(size_t)s * ENC_D + tid];
        a1 += w * eb[(size_t)s * ENC_D + tid + 256];
    }
    cxh1_cur[(size_t)b * 768 + tid]       = (bf16_t)a0;   // ctx cols [0,512)
    cxh1_cur[(size_t)b * 768 + tid + 256] = (bf16_t)a1;
}

// Fused LSTM cell: gate GEMM (4 gates per wave, shared A frag) + bias + activations + c/h update.
// Block = 16-wide u-chunk (gridDim.x = 16), wave = 16-row M tile (8 waves).
// All 8 waves consume identical B fragments, so the 4 gate tiles (4KB per K-step) are staged
// through LDS with double-buffered async global->LDS copies (ASYNCcnt path on CDNA5),
// cutting redundant L2 traffic 8x and overlapping the copy of kt+1 with the WMMAs of kt.
__global__ void lstm_kernel(const bf16_t* __restrict__ a0, int ld0, int klim0,
                            const bf16_t* __restrict__ a1, int ld1,
                            const bf16_t* __restrict__ Wp, int KT,
                            const float* __restrict__ bias,
                            float* __restrict__ cstate,
                            bf16_t* __restrict__ hout0, int hout0_ld, int hout0_off,
                            bf16_t* __restrict__ hout1, int hout1_ld, int hout1_off,
                            float* __restrict__ houtf) {
    constexpr int NT = 64;   // 1024 gate columns / 16
    __shared__ __align__(32) bf16_t btile[2][4 * 512];   // 2 stages x 4 gate tiles x 1KB
    int tid = threadIdx.x;
    int lane = tid & 31;
    int wave = tid >> 5;     // M tile 0..7
    int uc = blockIdx.x;     // u-chunk 0..15
    int row0 = wave * 16;

    // Stage copy: 256 threads x 16B = 4KB. chunk/gate = tid>>6, sub = tid&63.
    int gate = tid >> 6;
    int sub = tid & 63;
    int nt_g = uc + gate * 16;
    auto stage_fill = [&](int kt, int stage) {
        const bf16_t* gsrc = Wp + (((size_t)kt * NT + nt_g) * 32) * 16 + (size_t)sub * 8;
        bf16_t* ldst = &btile[stage][gate * 512 + sub * 8];
#if HAVE_ASYNC_LDS
        __builtin_amdgcn_global_load_async_to_lds_b128(
            (as1_v4i*)gsrc, (as3_v4i*)ldst, 0, 0);
#else
        *(u32x4*)ldst = *(const u32x4*)gsrc;
#endif
    };

    v8f ai = vzero8(), af = vzero8(), ag = vzero8(), ao = vzero8();

    stage_fill(0, 0);
#if HAVE_ASYNC_LDS
    async_wait0();
#endif
    __syncthreads();

    for (int kt = 0; kt < KT; ++kt) {
        int cur = kt & 1;
        if (kt + 1 < KT) stage_fill(kt + 1, cur ^ 1);

        int k0 = kt * 32;
        v16bf a = (k0 < klim0) ? load_a_bf16(a0, ld0, row0, k0, lane)
                               : load_a_bf16(a1, ld1, row0, k0 - klim0, lane);
        v16bf bi = *(const v16bf*)&btile[cur][0 * 512 + lane * 16];
        v16bf bf = *(const v16bf*)&btile[cur][1 * 512 + lane * 16];
        v16bf bg = *(const v16bf*)&btile[cur][2 * 512 + lane * 16];
        v16bf bo = *(const v16bf*)&btile[cur][3 * 512 + lane * 16];
        ai = wmma_bf16(a, bi, ai);
        af = wmma_bf16(a, bf, af);
        ag = wmma_bf16(a, bg, ag);
        ao = wmma_bf16(a, bo, ao);

#if HAVE_ASYNC_LDS
        async_wait0();            // stage kt+1 resident before the barrier releases it
#endif
        __syncthreads();          // also fences this wave's ds reads of stage kt
    }

    int u = uc * 16 + (lane & 15);
    float b_i = bias[u], b_f = bias[U + u], b_g = bias[2 * U + u], b_o = bias[3 * U + u];
    int rbase = row0 + ((lane >> 4) ? 8 : 0);
#pragma unroll
    for (int r = 0; r < 8; ++r) {
        int brow = rbase + r;
        float i = sigmoidf_(ai[r] + b_i);
        float f = sigmoidf_(af[r] + b_f);
        float g = fast_tanh(ag[r] + b_g);
        float o = sigmoidf_(ao[r] + b_o);
        size_t cidx = (size_t)brow * U + u;
        float cn = f * cstate[cidx] + i * g;
        cstate[cidx] = cn;
        float h = o * fast_tanh(cn);
        bf16_t hb = (bf16_t)h;
        hout0[(size_t)brow * hout0_ld + hout0_off + u] = hb;
        if (hout1) hout1[(size_t)brow * hout1_ld + hout1_off + u] = hb;
        if (houtf) houtf[cidx] = h;
    }
}

// pred[b, t] = h2 @ Wo + bo
__global__ void pred_kernel(const float* __restrict__ h2, const float* __restrict__ Wo,
                            const float* __restrict__ bo, float* __restrict__ out, int t) {
    int b = threadIdx.x;   // 128 threads
    float acc = 0.0f;
#pragma unroll 8
    for (int u = 0; u < U; ++u) acc += h2[b * U + u] * Wo[u];
    out[b * T + t] = acc + bo[0];
}

// ---- Host -------------------------------------------------------------------

extern "C" void kernel_launch(void* const* d_in, const int* in_sizes, int n_in,
                              void* d_out, int out_size, void* d_ws, size_t ws_size,
                              hipStream_t stream) {
    const float* x    = (const float*)d_in[0];
    const float* eh1  = (const float*)d_in[1];
    const float* ec1  = (const float*)d_in[2];
    const float* eh2  = (const float*)d_in[3];
    const float* ec2  = (const float*)d_in[4];
    const float* enc  = (const float*)d_in[5];
    const float* embW = (const float*)d_in[6];
    const float* embB = (const float*)d_in[7];
    const float* W1   = (const float*)d_in[8];
    const float* b1   = (const float*)d_in[9];
    const float* W2   = (const float*)d_in[10];
    const float* b2   = (const float*)d_in[11];
    const float* V    = (const float*)d_in[12];
    const float* bV   = (const float*)d_in[13];
    const float* Wx1  = (const float*)d_in[14];
    const float* Wh1  = (const float*)d_in[15];
    const float* bl1  = (const float*)d_in[16];
    const float* Wx2  = (const float*)d_in[17];
    const float* Wh2  = (const float*)d_in[18];
    const float* bl2  = (const float*)d_in[19];
    const float* Wo   = (const float*)d_in[20];
    const float* bo   = (const float*)d_in[21];
    float* out = (float*)d_out;

    char* ws = (char*)d_ws;
    size_t off = 0;
    auto alloc = [&](size_t bytes) -> char* {
        char* p = ws + off;
        off += (bytes + 255) & ~(size_t)255;
        return p;
    };
    float*  keys  = (float*) alloc((size_t)B * S * ATTN_U * 4);   // 16 MB
    bf16_t* W1p   = (bf16_t*)alloc(16 * 8 * 1024);
    bf16_t* W2p   = (bf16_t*)alloc(8 * 8 * 1024);
    bf16_t* Wxh1p = (bf16_t*)alloc((size_t)32 * 64 * 1024);       // 2 MB
    bf16_t* Wxh2p = (bf16_t*)alloc((size_t)16 * 64 * 1024);       // 1 MB
    bf16_t* embA  = (bf16_t*)alloc((size_t)T * B * EMB_D * 2);    // 3 MB
    bf16_t* cxh1a = (bf16_t*)alloc((size_t)B * 768 * 2);          // [ctx | h1] ping
    bf16_t* cxh1b = (bf16_t*)alloc((size_t)B * 768 * 2);          //            pong
    bf16_t* in2a  = (bf16_t*)alloc((size_t)B * 512 * 2);          // [h1n | h2] ping
    bf16_t* in2b  = (bf16_t*)alloc((size_t)B * 512 * 2);          //            pong
    float*  c1s   = (float*) alloc((size_t)B * U * 4);
    float*  c2s   = (float*) alloc((size_t)B * U * 4);
    float*  h2f   = (float*) alloc((size_t)B * U * 4);
    float*  qbuf  = (float*) alloc((size_t)B * ATTN_U * 4);

    auto nb = [](int n) { return (n + 255) / 256; };
    const int BIG = 1 << 30;

    // ---- prep (deterministic, rebuilt every call) ----
    pack_b_kernel<<<nb(16 * 8 * 32),  256, 0, stream>>>(W1, W1, BIG, 128, W1p, 16, 8);
    pack_b_kernel<<<nb(8 * 8 * 32),   256, 0, stream>>>(W2, W2, BIG, 128, W2p, 8, 8);
    pack_b_kernel<<<nb(32 * 64 * 32), 256, 0, stream>>>(Wx1, Wh1, 768, 1024, Wxh1p, 32, 64);
    pack_b_kernel<<<nb(16 * 64 * 32), 256, 0, stream>>>(Wx2, Wh2, 256, 1024, Wxh2p, 16, 64);
    emb_kernel<<<nb(T * B * EMB_D), 256, 0, stream>>>(x, embW, embB, embA);
    init_state_kernel<<<nb(B * U), 256, 0, stream>>>(eh1, ec1, eh2, ec2, cxh1a, in2a, c1s, c2s);
    keys_kernel<<<(B * S) / 16, 256, 0, stream>>>(enc, W1p, b1, keys);

    // ---- recurrent loop ----
    for (int t = 0; t < T; ++t) {
        bf16_t* cxh1_cur = (t & 1) ? cxh1b : cxh1a;
        bf16_t* cxh1_nxt = (t & 1) ? cxh1a : cxh1b;
        bf16_t* in2_cur  = (t & 1) ? in2b  : in2a;
        bf16_t* in2_nxt  = (t & 1) ? in2a  : in2b;

        qgemm_kernel<<<8, 256, 0, stream>>>(in2_cur + 256, W2p, b2, qbuf);
        attn_kernel<<<B, 256, 0, stream>>>(keys, qbuf, V, bV, enc, cxh1_cur);

        // LSTM1: A = [emb_t (K<256) | ctx,h1 (K>=256)], writes h1n -> in2_cur[:,0:256] and cxh1_nxt[:,512:768]
        lstm_kernel<<<16, 256, 0, stream>>>(embA + (size_t)t * B * EMB_D, EMB_D, 256,
                                            cxh1_cur, 768,
                                            Wxh1p, 32, bl1, c1s,
                                            in2_cur, 512, 0,
                                            cxh1_nxt, 768, 512,
                                            nullptr);
        // LSTM2: A = in2_cur = [h1n | h2old], writes h2n -> in2_nxt[:,256:512] and fp32 for pred
        lstm_kernel<<<16, 256, 0, stream>>>(in2_cur, 512, BIG,
                                            in2_cur, 512,
                                            Wxh2p, 16, bl2, c2s,
                                            in2_nxt, 512, 256,
                                            nullptr, 0, 0,
                                            h2f);
        pred_kernel<<<1, 128, 0, stream>>>(h2f, Wo, bo, out, t);
    }
}
